// _StackedAttnLayers_3289944948984
// MI455X (gfx1250) — matrix operand order repaired and checked by reading, not compiled, not run on hardware
//
#include <hip/hip_runtime.h>
#include <hip/hip_bf16.h>

typedef __bf16 bf16_t;
typedef __attribute__((ext_vector_type(16))) __bf16 v16bf;
typedef __attribute__((ext_vector_type(8)))  float  v8f;
typedef __attribute__((ext_vector_type(4)))  unsigned int v4u;

#define N_B     4
#define N_C     256
#define N_T     16
#define N_HW    196
#define N_THW   3136          // 16*196
#define DEMO_S  784           // 4*196
#define OBS_S   2352          // 12*196
#define N_HEADS 8
#define N_CH    32
#define INV_TEMP (1.0f/16.0f) // 1/sqrt(256)

// ---------------------------------------------------------------------------
// WMMA helpers (wave32, 16x16x32 bf16 -> f32)
// ---------------------------------------------------------------------------
__device__ __forceinline__ v8f wmma_bf16(v16bf a, v16bf b, v8f c) {
  return __builtin_amdgcn_wmma_f32_16x16x32_bf16(false, a, false, b, (short)0, c,
                                                 false, false);
}

union Frag {
  v4u u[2];
  unsigned int w[8];
  v16bf bf;
};

// A matrix 16x32 (MxK) stored as packed pairs: word [m*ldw + k2] = (A[m][2k2],A[m][2k2+1]).
__device__ __forceinline__ v16bf load_a_pack(const unsigned int* base, int ldw, int k2off) {
  int lane = threadIdx.x & 31;
  int half = lane >> 4, m = lane & 15;
  Frag f;
  f.u[0] = *(const v4u*)(base + m * ldw + k2off + 8 * half);
  f.u[1] = *(const v4u*)(base + m * ldw + k2off + 8 * half + 4);
  return f.bf;
}

// B matrix 32x16 (KxN) stored transposed-packed: word [n*ldw + k2] = (B[2k2][n],B[2k2+1][n]).
__device__ __forceinline__ v16bf load_b_packT(const unsigned int* base, int ldw) {
  int lane = threadIdx.x & 31;
  int half = lane >> 4, n = lane & 15;
  Frag f;
  f.u[0] = *(const v4u*)(base + n * ldw + 8 * half);
  f.u[1] = *(const v4u*)(base + n * ldw + 8 * half + 4);
  return f.bf;
}

// Single-instruction packed fp32->bf16x2 (low = a, high = b).
__device__ __forceinline__ unsigned int pack2f(float a, float b) {
  unsigned int r;
  asm("v_cvt_pk_bf16_f32 %0, %1, %2" : "=v"(r) : "v"(a), "v"(b));
  return r;
}
__device__ __forceinline__ unsigned int pack2h(bf16_t a, bf16_t b) {
  union { bf16_t h[2]; unsigned int u; } z;
  z.h[0] = a;
  z.h[1] = b;
  return z.u;
}

// CDNA5 async global->LDS copy (8 bytes per lane), tracked by ASYNCcnt.
__device__ __forceinline__ void async_g2l_b64(unsigned lds_off, const void* gptr) {
  asm volatile("global_load_async_to_lds_b64 %0, %1, off"
               :: "v"(lds_off), "v"(gptr)
               : "memory");
}
__device__ __forceinline__ void wait_async() {
  asm volatile("s_wait_asynccnt 0" ::: "memory");
}

// ---------------------------------------------------------------------------
// QKV projection: Y[b,o,s] = sum_c W[o,c] * X[b,c,toff+s], bf16 out
// grid: (ceil(S/128), 4, B*3), block 256.
// ---------------------------------------------------------------------------
__global__ __launch_bounds__(256)
void conv_qkv_kernel(const float* __restrict__ wq, const float* __restrict__ wk,
                     const float* __restrict__ wv, const float* __restrict__ x,
                     int toff, int S,
                     bf16_t* __restrict__ yq, bf16_t* __restrict__ yk,
                     bf16_t* __restrict__ yv) {
  __shared__ __align__(16) unsigned int sWp[64 * 24];   // A pack [m][k2]
  __shared__ __align__(16) unsigned int sXp[128 * 24];  // B packT [n][k2]
  int b = blockIdx.z / 3, sel = blockIdx.z % 3;
  const float* Wm = (sel == 0) ? wq : (sel == 1) ? wk : wv;
  bf16_t* Y       = (sel == 0) ? yq : (sel == 1) ? yk : yv;
  int mbase = blockIdx.y * 64;
  int nbase = blockIdx.x * 128;
  int tid = threadIdx.x, wave = tid >> 5, lane = tid & 31;
  int half = lane >> 4, nn = lane & 15;
  int wm = (wave & 3) * 16;
  int wn = (wave >> 2) * 64;
  const float* xb = x + (size_t)b * N_C * N_THW + toff;
  v8f acc[4];
#pragma unroll
  for (int j = 0; j < 4; ++j) acc[j] = (v8f){0.f,0.f,0.f,0.f,0.f,0.f,0.f,0.f};

  for (int k0 = 0; k0 < N_C; k0 += 32) {
#pragma unroll
    for (int it = 0; it < 4; ++it) {
      int idx = it * 256 + tid;
      int k2 = idx & 15, r = idx >> 4;
      const float* wp = Wm + (size_t)(mbase + r) * N_C + k0 + 2 * k2;
      sWp[r * 24 + k2] = pack2f(wp[0], wp[1]);
    }
#pragma unroll
    for (int it = 0; it < 8; ++it) {
      int idx = it * 256 + tid;
      int n = idx & 127, k2 = idx >> 7;
      int s = nbase + n;
      float v0 = 0.f, v1 = 0.f;
      if (s < S) {
        const float* xp = xb + (size_t)(k0 + 2 * k2) * N_THW + s;
        v0 = xp[0];
        v1 = xp[N_THW];
      }
      sXp[n * 24 + k2] = pack2f(v0, v1);
    }
    if (k0 + 32 < N_C) {
      __builtin_prefetch(Wm + (size_t)(mbase + (tid >> 2)) * N_C + k0 + 32, 0, 0);
      __builtin_prefetch(xb + (size_t)(k0 + 32 + (tid >> 4)) * N_THW + nbase, 0, 0);
    }
    __syncthreads();
    v16bf a = load_a_pack(sWp + wm * 24, 24, 0);
#pragma unroll
    for (int j = 0; j < 4; ++j) {
      v16bf bb = load_b_packT(sXp + (wn + 16 * j) * 24, 24);
      acc[j] = wmma_bf16(a, bb, acc[j]);
    }
    __syncthreads();
  }
#pragma unroll
  for (int j = 0; j < 4; ++j)
#pragma unroll
    for (int r = 0; r < 8; ++r) {
      int o = mbase + wm + r + 8 * half;
      int s = nbase + wn + 16 * j + nn;
      if (s < S) Y[((size_t)b * N_C + o) * S + s] = (bf16_t)acc[j][r];
    }
}

// ---------------------------------------------------------------------------
// Output conv + residual + relu into x obs region (in place).
// ---------------------------------------------------------------------------
__global__ __launch_bounds__(256)
void conv_out_kernel(const float* __restrict__ Wm,
                     const float* __restrict__ merged,
                     float* __restrict__ xcur) {
  __shared__ __align__(16) unsigned int sWp[64 * 24];
  __shared__ __align__(16) unsigned int sXp[128 * 24];
  int b = blockIdx.z;
  int mbase = blockIdx.y * 64;
  int nbase = blockIdx.x * 128;
  int tid = threadIdx.x, wave = tid >> 5, lane = tid & 31;
  int half = lane >> 4, nn = lane & 15;
  int wm = (wave & 3) * 16;
  int wn = (wave >> 2) * 64;
  const float* xb = merged + (size_t)b * N_C * OBS_S;
  v8f acc[4];
#pragma unroll
  for (int j = 0; j < 4; ++j) acc[j] = (v8f){0.f,0.f,0.f,0.f,0.f,0.f,0.f,0.f};

  for (int k0 = 0; k0 < N_C; k0 += 32) {
#pragma unroll
    for (int it = 0; it < 4; ++it) {
      int idx = it * 256 + tid;
      int k2 = idx & 15, r = idx >> 4;
      const float* wp = Wm + (size_t)(mbase + r) * N_C + k0 + 2 * k2;
      sWp[r * 24 + k2] = pack2f(wp[0], wp[1]);
    }
#pragma unroll
    for (int it = 0; it < 8; ++it) {
      int idx = it * 256 + tid;
      int n = idx & 127, k2 = idx >> 7;
      int s = nbase + n;
      float v0 = 0.f, v1 = 0.f;
      if (s < OBS_S) {
        const float* xp = xb + (size_t)(k0 + 2 * k2) * OBS_S + s;
        v0 = xp[0];
        v1 = xp[OBS_S];
      }
      sXp[n * 24 + k2] = pack2f(v0, v1);
    }
    __syncthreads();
    v16bf a = load_a_pack(sWp + wm * 24, 24, 0);
#pragma unroll
    for (int j = 0; j < 4; ++j) {
      v16bf bb = load_b_packT(sXp + (wn + 16 * j) * 24, 24);
      acc[j] = wmma_bf16(a, bb, acc[j]);
    }
    __syncthreads();
  }
#pragma unroll
  for (int j = 0; j < 4; ++j)
#pragma unroll
    for (int r = 0; r < 8; ++r) {
      int o = mbase + wm + r + 8 * half;
      int s = nbase + wn + 16 * j + nn;
      if (s < OBS_S) {
        float* xo = xcur + ((size_t)b * N_C + o) * N_THW + DEMO_S;
        xo[s] = xo[s] + fmaxf(acc[j][r], 0.f);  // residual + relu, in place
      }
    }
}

// ---------------------------------------------------------------------------
// Demo attention, fused: per (i-block of 64, head, b).
// Dynamic LDS: sS 201728 | sKAp 6144 | sQp/sPp (aliased) 102400 = 310,272 B
// ---------------------------------------------------------------------------
__global__ __launch_bounds__(256)
void demo_attn_kernel(const bf16_t* __restrict__ dq, const bf16_t* __restrict__ dk,
                      const bf16_t* __restrict__ dv, bf16_t* __restrict__ dva) {
  extern __shared__ char smem[];
  float*        sS   = (float*)smem;                          // [64][788]
  unsigned int* sKAp = (unsigned int*)(smem + 201728);        // [64][24]
  unsigned int* sQp  = (unsigned int*)(smem + 207872);        // [784][24] packT
  unsigned int* sPp  = sQp;                                   // [64][400] pack (alias)

  const int ibase = blockIdx.x * 64;
  const int nh = blockIdx.y, b = blockIdx.z;
  const int tid = threadIdx.x, wave = tid >> 5, lane = tid & 31;
  const int half = lane >> 4, nn = lane & 15;
  const size_t hb = (size_t)b * N_C + nh * N_CH;

  // stage K^T block (A pack: rows i_local, pairs along c)
#pragma unroll
  for (int it = 0; it < 4; ++it) {
    int idx = it * 256 + tid;
    int r = idx & 63, k2 = idx >> 6;
    int i = ibase + r;
    bf16_t v0 = (bf16_t)0.f, v1 = (bf16_t)0.f;
    if (i < DEMO_S) {
      v0 = dk[(hb + 2 * k2) * DEMO_S + i];
      v1 = dk[(hb + 2 * k2 + 1) * DEMO_S + i];
    }
    sKAp[r * 24 + k2] = pack2h(v0, v1);
  }
  // stage whole Q (B packT: [j][c/2])
  {
    const bf16_t* qb = dq + hb * DEMO_S;
#pragma unroll 7
    for (int it = 0; it < 49; ++it) {
      int idx = it * 256 + tid;
      int k2 = idx / DEMO_S;
      int n = idx - k2 * DEMO_S;
      sQp[n * 24 + k2] = pack2h(qb[(size_t)(2 * k2) * DEMO_S + n],
                                qb[(size_t)(2 * k2 + 1) * DEMO_S + n]);
    }
  }
  __syncthreads();

  v16bf afr[4];
#pragma unroll
  for (int mi = 0; mi < 4; ++mi) afr[mi] = load_a_pack(sKAp + (mi * 16) * 24, 24, 0);

  // Phase 1: scores (49 j-tiles x 4 m-tiles over 8 waves)
  for (int jt = wave; jt < 49; jt += 8) {
    int jbase = jt * 16;
    v16bf bq = load_b_packT(sQp + jbase * 24, 24);
#pragma unroll
    for (int mi = 0; mi < 4; ++mi) {
      v8f z = (v8f){0.f,0.f,0.f,0.f,0.f,0.f,0.f,0.f};
      v8f d = wmma_bf16(afr[mi], bq, z);
#pragma unroll
      for (int r = 0; r < 8; ++r)
        sS[(mi * 16 + r + 8 * half) * 788 + jbase + nn] = d[r];
    }
  }
  __syncthreads();

  // Phase 2: row softmax over j; emit packed bf16 P into sPp (aliases sQp)
  for (int r = wave; r < 64; r += 8) {
    float mx = -3.0e38f;
    for (int j = lane; j < DEMO_S; j += 32) mx = fmaxf(mx, sS[r * 788 + j]);
#pragma unroll
    for (int off = 16; off >= 1; off >>= 1) mx = fmaxf(mx, __shfl_xor(mx, off, 32));
    mx *= INV_TEMP;
    float sum = 0.f;
    for (int j = lane; j < DEMO_S; j += 32) {
      float e = __expf(sS[r * 788 + j] * INV_TEMP - mx);
      sS[r * 788 + j] = e;
      sum += e;
    }
#pragma unroll
    for (int off = 16; off >= 1; off >>= 1) sum += __shfl_xor(sum, off, 32);
    float inv = 1.f / sum;
    for (int j0 = lane * 2; j0 < 800; j0 += 64) {
      float e0 = (j0 < DEMO_S) ? sS[r * 788 + j0] * inv : 0.f;
      float e1 = (j0 + 1 < DEMO_S) ? sS[r * 788 + j0 + 1] * inv : 0.f;
      sPp[r * 400 + (j0 >> 1)] = pack2f(e0, e1);
    }
  }
  __syncthreads();

  // Phase 3: O[i_local, c] = sum_j P[i,j] V[c,j].  8 tiles -> 1 per wave.
  int mi = wave & 3;
  int cj = wave >> 2;  // 16-channel half
  const unsigned int* vb32 =
      (const unsigned int*)(dv + (hb + cj * 16 + nn) * DEMO_S);
  v8f acc = (v8f){0.f,0.f,0.f,0.f,0.f,0.f,0.f,0.f};
#pragma unroll 5
  for (int kk = 0; kk < 25; ++kk) {
    v16bf a = load_a_pack(sPp + (mi * 16) * 400, 400, kk * 16);
    Frag fb;
#pragma unroll
    for (int v = 0; v < 8; ++v) {
      int k2 = kk * 16 + 8 * half + v;  // pair (2k2, 2k2+1) contiguous in global
      fb.w[v] = (2 * k2 < DEMO_S) ? vb32[k2] : 0u;
    }
    acc = wmma_bf16(a, fb.bf, acc);
  }
#pragma unroll
  for (int r = 0; r < 8; ++r) {
    int i = ibase + mi * 16 + r + 8 * half;
    if (i < DEMO_S) dva[(hb + cj * 16 + nn) * DEMO_S + i] = (bf16_t)acc[r];
  }
}

// ---------------------------------------------------------------------------
// Obs attention, flash-style over s (980 keys = 784 demo + 196 obs).
// Dynamic LDS: sQp 19968 | sS 26624 | sPp 19968 | sKAp 3072 | sVAp 3072 = 72704 B
// Tiles (2 c-halves x 13 p-tiles = 26): wave owns tl = wave + 8q, q = 0..3
// (compile-time unrolled so acc[] stays in VGPRs).
// ---------------------------------------------------------------------------
__global__ __launch_bounds__(256)
void obs_attn_kernel(const bf16_t* __restrict__ oq, const bf16_t* __restrict__ okb,
                     const bf16_t* __restrict__ ovb, const bf16_t* __restrict__ dk,
                     const bf16_t* __restrict__ dva, float* __restrict__ merged) {
  extern __shared__ char smem[];
  unsigned int* sQp  = (unsigned int*)smem;            // [208][24] packT (B: k=c)
  float*        sS   = (float*)(smem + 19968);         // [32][208]
  unsigned int* sPp  = (unsigned int*)(smem + 46592);  // [208][24] packT (B: k=s)
  unsigned int* sKAp = (unsigned int*)(smem + 66560);  // [32][24]  A pack (m=s,k=c)
  unsigned int* sVAp = (unsigned int*)(smem + 69632);  // [32][24]  A pack (m=c,k=s)

  const int nh = blockIdx.x, t = blockIdx.y, b = blockIdx.z;
  const int tid = threadIdx.x, wave = tid >> 5, lane = tid & 31;
  const int half = lane >> 4, nn = lane & 15;
  const size_t hb = (size_t)b * N_C + nh * N_CH;
  const size_t obs_off = (size_t)t * N_HW;

  // stage Q packT: [p][c/2]
#pragma unroll
  for (int it = 0; it < 13; ++it) {
    int idx = it * 256 + tid;
    int k2 = idx / 208;
    int n = idx - k2 * 208;
    unsigned int w = 0;
    if (n < N_HW)
      w = pack2h(oq[(hb + 2 * k2) * OBS_S + obs_off + n],
                 oq[(hb + 2 * k2 + 1) * OBS_S + obs_off + n]);
    sQp[n * 24 + k2] = w;
  }

  v8f acc[4];
#pragma unroll
  for (int q = 0; q < 4; ++q) acc[q] = (v8f){0.f,0.f,0.f,0.f,0.f,0.f,0.f,0.f};
  __syncthreads();

  for (int sc = 0; sc < 31; ++sc) {  // 980 keys padded to 992
    int sbase = sc * 32;
    // K chunk: A pack [s_local][c/2]
#pragma unroll
    for (int it = 0; it < 2; ++it) {
      int idx = it * 256 + tid;
      int sl = idx & 31, k2 = idx >> 5;
      int s = sbase + sl;
      bf16_t k0 = (bf16_t)0.f, k1 = (bf16_t)0.f;
      if (s < DEMO_S) {
        k0 = dk[(hb + 2 * k2) * DEMO_S + s];
        k1 = dk[(hb + 2 * k2 + 1) * DEMO_S + s];
      } else if (s < DEMO_S + N_HW) {
        k0 = okb[(hb + 2 * k2) * OBS_S + obs_off + (s - DEMO_S)];
        k1 = okb[(hb + 2 * k2 + 1) * OBS_S + obs_off + (s - DEMO_S)];
      }
      sKAp[sl * 24 + k2] = pack2h(k0, k1);
    }
    // V chunk: A pack [c][s_local/2].  Verbatim global->LDS copy via the
    // CDNA5 async-to-LDS engine (8B per lane; 4-s-value segments never
    // straddle the 784/980 boundaries; all addresses 8B aligned).
    {
      int c = tid >> 3, g = tid & 7;
      int s0 = sbase + 4 * g;
      unsigned int* dst = sVAp + c * 24 + 2 * g;
      unsigned lds_off = (unsigned)(size_t)dst;
      if (s0 < DEMO_S) {
        async_g2l_b64(lds_off, dva + (hb + c) * DEMO_S + s0);
      } else if (s0 < DEMO_S + N_HW) {
        async_g2l_b64(lds_off, ovb + (hb + c) * OBS_S + obs_off + (s0 - DEMO_S));
      } else {
        *(unsigned long long*)dst = 0ull;  // zero pad s >= 980
      }
    }
    wait_async();
    __syncthreads();

    // scores: S[s_local, p] = sum_c K[c,s] Q[c,p]
#pragma unroll
    for (int q = 0; q < 4; ++q) {
      int tl = wave + q * 8;
      if (tl < 26) {
        int mi = tl & 1, ni = tl >> 1;
        v16bf a = load_a_pack(sKAp + (mi * 16) * 24, 24, 0);
        v16bf bq = load_b_packT(sQp + (ni * 16) * 24, 24);
        v8f z = (v8f){0.f,0.f,0.f,0.f,0.f,0.f,0.f,0.f};
        v8f d = wmma_bf16(a, bq, z);
#pragma unroll
        for (int r = 0; r < 8; ++r)
          sS[(mi * 16 + r + 8 * half) * 208 + ni * 16 + nn] = d[r];
      }
    }
    __syncthreads();

    // softmax over p for the 32 s-rows; rows processed in pairs so P can be
    // written directly in packT ([p][s/2]) layout.
#pragma unroll
    for (int pr = 0; pr < 2; ++pr) {
      int r0 = wave * 4 + pr * 2, r1 = r0 + 1;
      float mx0 = -3.0e38f, mx1 = -3.0e38f;
      for (int j = lane; j < N_HW; j += 32) {
        mx0 = fmaxf(mx0, sS[r0 * 208 + j]);
        mx1 = fmaxf(mx1, sS[r1 * 208 + j]);
      }
#pragma unroll
      for (int off = 16; off >= 1; off >>= 1) {
        mx0 = fmaxf(mx0, __shfl_xor(mx0, off, 32));
        mx1 = fmaxf(mx1, __shfl_xor(mx1, off, 32));
      }
      mx0 *= INV_TEMP;
      mx1 *= INV_TEMP;
      float s0 = 0.f, s1 = 0.f;
      for (int j = lane; j < N_HW; j += 32) {
        float e0 = __expf(sS[r0 * 208 + j] * INV_TEMP - mx0);
        float e1 = __expf(sS[r1 * 208 + j] * INV_TEMP - mx1);
        sS[r0 * 208 + j] = e0;
        sS[r1 * 208 + j] = e1;
        s0 += e0;
        s1 += e1;
      }
#pragma unroll
      for (int off = 16; off >= 1; off >>= 1) {
        s0 += __shfl_xor(s0, off, 32);
        s1 += __shfl_xor(s1, off, 32);
      }
      float inv0 = 1.f / s0, inv1 = 1.f / s1;
      int k2 = r0 >> 1;
      for (int p = lane; p < 208; p += 32) {
        float e0 = (p < N_HW) ? sS[r0 * 208 + p] * inv0 : 0.f;
        float e1 = (p < N_HW) ? sS[r1 * 208 + p] * inv1 : 0.f;
        sPp[p * 24 + k2] = pack2f(e0, e1);
      }
    }
    __syncthreads();

    // out += V_chunk (32c x 32s) * P_chunk (32s x 208p)
#pragma unroll
    for (int q = 0; q < 4; ++q) {
      int tl = wave + q * 8;
      if (tl < 26) {
        int mi = tl & 1, ni = tl >> 1;
        v16bf a = load_a_pack(sVAp + (mi * 16) * 24, 24, 0);
        v16bf bp = load_b_packT(sPp + (ni * 16) * 24, 24);
        acc[q] = wmma_bf16(a, bp, acc[q]);
      }
    }
    __syncthreads();
  }

#pragma unroll
  for (int q = 0; q < 4; ++q) {
    int tl = wave + q * 8;
    if (tl < 26) {
      int mi = tl & 1, ni = tl >> 1;
#pragma unroll
      for (int r = 0; r < 8; ++r) {
        int c = mi * 16 + r + 8 * half;
        int p = ni * 16 + nn;
        if (p < N_HW) merged[(hb + c) * OBS_S + obs_off + p] = acc[q][r];
      }
    }
  }
}

// ---------------------------------------------------------------------------
// BatchNorm: per-channel mean/var over (B,T,H,W) then normalize.
// ---------------------------------------------------------------------------
__global__ __launch_bounds__(256)
void bn_reduce_kernel(const float* __restrict__ x, float* __restrict__ mv) {
  int o = blockIdx.x;
  int tid = threadIdx.x;
  float s1 = 0.f, s2 = 0.f;
  for (int b = 0; b < N_B; ++b) {
    const float* p = x + ((size_t)b * N_C + o) * N_THW;
    for (int i = tid; i < N_THW; i += 256) {
      float v = p[i];
      s1 += v;
      s2 += v * v;
    }
  }
  __shared__ float r1[256], r2[256];
  r1[tid] = s1;
  r2[tid] = s2;
  __syncthreads();
  for (int off = 128; off >= 1; off >>= 1) {
    if (tid < off) { r1[tid] += r1[tid + off]; r2[tid] += r2[tid + off]; }
    __syncthreads();
  }
  if (tid == 0) {
    float n = (float)(N_B * N_THW);
    float mean = r1[0] / n;
    float var = r2[0] / n - mean * mean;
    mv[o * 2] = mean;
    mv[o * 2 + 1] = var;
  }
}

__global__ __launch_bounds__(256)
void bn_norm_kernel(const float* __restrict__ x, const float* __restrict__ mv,
                    const float* __restrict__ gamma, const float* __restrict__ beta,
                    float* __restrict__ y) {
  size_t idx = (size_t)blockIdx.x * 256 + threadIdx.x;
  size_t total = (size_t)N_B * N_C * N_THW;
  if (idx >= total) return;
  int o = (int)((idx / N_THW) & (N_C - 1));
  float mean = mv[o * 2], var = mv[o * 2 + 1];
  y[idx] = gamma[o] * (x[idx] - mean) * rsqrtf(var + 1e-5f) + beta[o];
}

// ---------------------------------------------------------------------------
// Orchestration
// ---------------------------------------------------------------------------
extern "C" void kernel_launch(void* const* d_in, const int* in_sizes, int n_in,
                              void* d_out, int out_size, void* d_ws, size_t ws_size,
                              hipStream_t stream) {
  const float* xin = (const float*)d_in[0];
  const float* dqw = (const float*)d_in[1];
  const float* dkw = (const float*)d_in[2];
  const float* dvw = (const float*)d_in[3];
  const float* oqw = (const float*)d_in[4];
  const float* okw = (const float*)d_in[5];
  const float* ovw = (const float*)d_in[6];
  const float* oww = (const float*)d_in[7];
  const float* gam = (const float*)d_in[8];
  const float* bet = (const float*)d_in[9];

  char* ws = (char*)d_ws;
  size_t off = 0;
  auto alloc = [&](size_t bytes) -> void* {
    void* p = ws + off;
    off = (off + bytes + 255) & ~(size_t)255;
    return p;
  };
  const size_t xElems = (size_t)N_B * N_C * N_THW;      // 3,211,264
  const size_t dElems = (size_t)N_B * N_C * DEMO_S;     // 802,816
  const size_t oElems = (size_t)N_B * N_C * OBS_S;      // 2,408,448

  float*  xcur   = (float*)alloc(xElems * 4);
  bf16_t* dq     = (bf16_t*)alloc(dElems * 2);
  bf16_t* dk     = (bf16_t*)alloc(dElems * 2);
  bf16_t* dv     = (bf16_t*)alloc(dElems * 2 + 64);
  bf16_t* dva    = (bf16_t*)alloc(dElems * 2 + 64);
  bf16_t* oq     = (bf16_t*)alloc(oElems * 2);
  bf16_t* ok     = (bf16_t*)alloc(oElems * 2);
  bf16_t* ov     = (bf16_t*)alloc(oElems * 2 + 64);
  float*  merged = (float*)alloc(oElems * 4);
  float*  mv     = (float*)alloc(512 * 4);

  hipMemcpyAsync(xcur, xin, xElems * 4, hipMemcpyDeviceToDevice, stream);

  const size_t demo_dyn = 201728 + 6144 + 102400;               // 310,272 B
  const size_t obs_dyn  = 19968 + 26624 + 19968 + 3072 + 3072;  // 72,704 B

  for (int i = 0; i < 3; ++i) {
    size_t woff = (size_t)i * N_C * N_C;
    conv_qkv_kernel<<<dim3(7, 4, N_B * 3), 256, 0, stream>>>(
        dqw + woff, dkw + woff, dvw + woff, xcur, 0, DEMO_S, dq, dk, dv);
    conv_qkv_kernel<<<dim3(19, 4, N_B * 3), 256, 0, stream>>>(
        oqw + woff, okw + woff, ovw + woff, xcur, DEMO_S, OBS_S, oq, ok, ov);
    demo_attn_kernel<<<dim3(13, N_HEADS, N_B), 256, demo_dyn, stream>>>(
        dq, dk, dv, dva);
    obs_attn_kernel<<<dim3(N_HEADS, 12, N_B), 256, obs_dyn, stream>>>(
        oq, ok, ov, dk, dva, merged);
    conv_out_kernel<<<dim3(19, 4, N_B), 256, 0, stream>>>(
        oww + woff, merged, xcur);
    bn_reduce_kernel<<<dim3(N_C), 256, 0, stream>>>(xcur, mv);
    float* dst = (i == 2) ? (float*)d_out : xcur;
    bn_norm_kernel<<<dim3((unsigned)((xElems + 255) / 256)), 256, 0, stream>>>(
        xcur, mv, gam + (size_t)i * N_C, bet + (size_t)i * N_C, dst);
  }
}
